// MultiHeadAttention_68925635166225
// MI455X (gfx1250) — matrix operand order repaired and checked by reading, not compiled, not run on hardware
//
#include <hip/hip_runtime.h>
#include <hip/hip_bf16.h>

// ---------- problem constants (from reference) ----------
#define Bb 4
#define Ss 2048
#define Dd 1024
#define Hh 16
#define DKk 64                 // D / H
#define BS (Bb * Ss)           // 8192 token rows

typedef __bf16 v16bf __attribute__((ext_vector_type(16)));
typedef float  v8f   __attribute__((ext_vector_type(8)));
typedef int    v4i   __attribute__((ext_vector_type(4)));

// async-to-LDS pointer casts: builtin wants (global v4i*, local v4i*, i32, i32)
#define ASYNC_G(p) ((__attribute__((address_space(1))) v4i*)(p))
#define ASYNC_L(p) ((__attribute__((address_space(3))) v4i*)(p))

// ======================================================================
// fp32 -> bf16 conversion (vectorized by 4)
// ======================================================================
__global__ void cvt_f32_bf16(const float* __restrict__ in,
                             __bf16* __restrict__ out, int n) {
  int i = (blockIdx.x * blockDim.x + threadIdx.x) * 4;
  if (i < n) {
    float4 f = *(const float4*)(in + i);
    out[i + 0] = (__bf16)f.x;
    out[i + 1] = (__bf16)f.y;
    out[i + 2] = (__bf16)f.z;
    out[i + 3] = (__bf16)f.w;
  }
}

// ======================================================================
// butterfly max over the 16-lane column group (DPP ROW_XMASK, pure VALU)
// ======================================================================
__device__ __forceinline__ float xmax16(float x) {
  int v;
  v = __builtin_amdgcn_update_dpp(0, __builtin_bit_cast(int, x), 0x161, 0xf, 0xf, true);
  x = fmaxf(x, __builtin_bit_cast(float, v));
  v = __builtin_amdgcn_update_dpp(0, __builtin_bit_cast(int, x), 0x162, 0xf, 0xf, true);
  x = fmaxf(x, __builtin_bit_cast(float, v));
  v = __builtin_amdgcn_update_dpp(0, __builtin_bit_cast(int, x), 0x164, 0xf, 0xf, true);
  x = fmaxf(x, __builtin_bit_cast(float, v));
  v = __builtin_amdgcn_update_dpp(0, __builtin_bit_cast(int, x), 0x168, 0xf, 0xf, true);
  x = fmaxf(x, __builtin_bit_cast(float, v));
  return x;
}

// ======================================================================
// GEMM: Y[M x 1024] = X[M x 1024] @ W[1024 x 1024]^T + bias
// One wave -> 16(M) x 64(N) tile, 4 f32 accumulators, K-loop of 32.
// ======================================================================
template <bool F32OUT>
__global__ __launch_bounds__(128) void gemm_xwT(
    const __bf16* __restrict__ X, const __bf16* __restrict__ W,
    const float* __restrict__ bias, void* __restrict__ Yv, int M) {
  const int wid  = (blockIdx.x * blockDim.x + threadIdx.x) >> 5;
  const int lane = threadIdx.x & 31;
  const int nTiles = Dd / 64;                 // 16
  const int mTile  = wid / nTiles;
  const int nTile  = wid % nTiles;
  if (mTile * 16 >= M) return;                // wave-uniform

  const int l15  = lane & 15;
  const int half = lane >> 4;
  const int aK   = half ? 8 : 0;              // A-frag K sub-offset
  const int bK   = half ? 8 : 0;              // B-frag K sub-offset
  const size_t arow = (size_t)(mTile * 16 + l15) * Dd;
  const int    wrow = nTile * 64 + l15;

  v8f acc[4] = {};

  for (int k = 0; k < Dd; k += 32) {
    v16bf a;
    *(uint4*)&a       = *(const uint4*)(X + arow + k + aK);
    *((uint4*)&a + 1) = *(const uint4*)(X + arow + k + 16 + aK);
#pragma unroll
    for (int nc = 0; nc < 4; ++nc) {
      const __bf16* wp = W + (size_t)(wrow + nc * 16) * Dd + k + bK;
      v16bf b;
      *(uint4*)&b       = *(const uint4*)(wp);
      *((uint4*)&b + 1) = *(const uint4*)(wp + 16);
      acc[nc] = __builtin_amdgcn_wmma_f32_16x16x32_bf16(
          false, a, false, b, (short)0, acc[nc], false, false);
    }
  }

#pragma unroll
  for (int nc = 0; nc < 4; ++nc) {
#pragma unroll
    for (int r = 0; r < 8; ++r) {
      const size_t mrow = (size_t)(mTile * 16 + r + half * 8);
      const int    col  = nTile * 64 + nc * 16 + l15;
      const float  v    = acc[nc][r] + bias[col];
      if (F32OUT) ((float*)Yv)[mrow * Dd + col] = v;
      else        ((__bf16*)Yv)[mrow * Dd + col] = (__bf16)v;
    }
  }
}

// ======================================================================
// Flash attention, block-cooperative:
//   block = 4 waves, one (b,h), 4 adjacent 16-query tiles (64 queries).
//   K/V tiles (32 keys) staged ONCE per block into shared LDS via
//   GLOBAL_LOAD_ASYNC_TO_LDS_B128 (ASYNCcnt), V transposed cooperatively.
//   Row max:  DPP ROW_XMASK butterflies (no LDS).
//   Row sum:  extra WMMA against an all-ones B fragment (free in C-layout).
// ======================================================================
#define KT_STRIDE 80   // padded row (bf16 elems) for K / V tiles, 160B (16B-aligned)
#define VT_STRIDE 48   // padded row for transposed V, 96B (16B-aligned)

__global__ __launch_bounds__(128) void flash_attn(
    const __bf16* __restrict__ Qp, const __bf16* __restrict__ Kp,
    const __bf16* __restrict__ Vp, __bf16* __restrict__ Ctx) {
  __shared__ __align__(16) __bf16 Kt[32 * KT_STRIDE];   // 32 keys x 64 dims
  __shared__ __align__(16) __bf16 Vl[32 * KT_STRIDE];   // 32 keys x 64 dims
  __shared__ __align__(16) __bf16 Vt[64 * VT_STRIDE];   // 64 dims x 32 keys
  __shared__ __align__(16) __bf16 ldsP[4][16 * 32];     // per-wave P tile

  const int w    = threadIdx.x >> 5;
  const int lane = threadIdx.x & 31;

  const int qBlocks = Ss / 64;                          // 32
  const int b  = blockIdx.x / (Hh * qBlocks);
  const int h  = (blockIdx.x / qBlocks) % Hh;
  const int qb = (blockIdx.x % qBlocks) * 64;           // block's first query
  const int qBase = qb + w * 16;                        // this wave's 16 queries

  const size_t headOff = (size_t)b * Ss * Dd + (size_t)h * DKk;
  const int l15  = lane & 15;
  const int half = lane >> 4;
  const int aK   = half ? 8 : 0;
  const int bK   = half ? 8 : 0;

  // ---- Q A-fragments (2 K-chunks of 32 dims), persistent in VGPRs ----
  v16bf qf[2];
  {
    const __bf16* qrow = Qp + headOff + (size_t)(qBase + l15) * Dd;
#pragma unroll
    for (int c = 0; c < 2; ++c) {
      *(uint4*)&qf[c]       = *(const uint4*)(qrow + c * 32 + aK);
      *((uint4*)&qf[c] + 1) = *(const uint4*)(qrow + c * 32 + 16 + aK);
    }
  }

  // all-ones B fragment: wmma(P, ones) -> row sums of P in every column
  v16bf ones;
#pragma unroll
  for (int i = 0; i < 16; ++i) ones[i] = (__bf16)1.0f;

  v8f acc[4] = {};
  v8f accL   = {};
  float mrow[8];
#pragma unroll
  for (int r = 0; r < 8; ++r) mrow[r] = -1e30f;

  const float scale = 0.125f;                           // DK^-0.5, DK=64
  const int nKt = qb / 32 + 2;                          // tiles up to qb+63

  for (int kt = 0; kt < nKt; ++kt) {
    const int kBase = kt * 32;

    // ---- async-stage K and V tiles into shared LDS (128 lanes coop) ----
    {
      const int t = threadIdx.x;
#pragma unroll
      for (int c = 0; c < 2; ++c) {
        const int i   = c * 128 + t;                    // 256 16B chunks/tile
        const int key = i >> 3;
        const int seg = (i & 7) * 8;                    // element offset
        const __bf16* gk = Kp + headOff + (size_t)(kBase + key) * Dd + seg;
        const __bf16* gv = Vp + headOff + (size_t)(kBase + key) * Dd + seg;
        __bf16* lk = &Kt[key * KT_STRIDE + seg];
        __bf16* lv = &Vl[key * KT_STRIDE + seg];
#if __has_builtin(__builtin_amdgcn_global_load_async_to_lds_b128)
        __builtin_amdgcn_global_load_async_to_lds_b128(ASYNC_G(gk), ASYNC_L(lk),
                                                       0, 0);
        __builtin_amdgcn_global_load_async_to_lds_b128(ASYNC_G(gv), ASYNC_L(lv),
                                                       0, 0);
#else
        *(uint4*)lk = *(const uint4*)gk;
        *(uint4*)lv = *(const uint4*)gv;
#endif
      }
#if __has_builtin(__builtin_amdgcn_global_load_async_to_lds_b128)
#if __has_builtin(__builtin_amdgcn_s_wait_asynccnt)
      __builtin_amdgcn_s_wait_asynccnt(0);
#else
      asm volatile("s_wait_asynccnt 0x0" ::: "memory");
#endif
#endif
    }
    __syncthreads();

    // ---- cooperative transpose V tile -> Vt[dim][key] ----
    {
      const int key = threadIdx.x & 31;
      const int dc  = threadIdx.x >> 5;                 // 16-dim chunk
      const __bf16* src = &Vl[key * KT_STRIDE + dc * 16];
      uint4 p0 = *(const uint4*)(src);
      uint4 p1 = *(const uint4*)(src + 8);
      const __bf16* e0 = (const __bf16*)&p0;
      const __bf16* e1 = (const __bf16*)&p1;
#pragma unroll
      for (int j = 0; j < 8; ++j) {
        Vt[(dc * 16 + j)     * VT_STRIDE + key] = e0[j];
        Vt[(dc * 16 + 8 + j) * VT_STRIDE + key] = e1[j];
      }
    }
    __syncthreads();

    // ---- compute (wave-uniform causal guard keeps EXEC all-ones) ----
    if (kBase <= qBase + 15) {
      // scores: two 16-key sub-tiles, K-chained over dk=64
      v8f s[2];
#pragma unroll
      for (int ns = 0; ns < 2; ++ns) {
        const __bf16* krow = &Kt[(ns * 16 + l15) * KT_STRIDE];
        v8f c = {};
#pragma unroll
        for (int ck = 0; ck < 2; ++ck) {
          v16bf kf;
          *(uint4*)&kf       = *(const uint4*)(krow + ck * 32 + bK);
          *((uint4*)&kf + 1) = *(const uint4*)(krow + ck * 32 + bK + 16);
          c = __builtin_amdgcn_wmma_f32_16x16x32_bf16(
              false, qf[ck], false, kf, (short)0, c, false, false);
        }
        s[ns] = c;
      }

      // scale + causal mask + row max (DPP butterflies, no LDS)
      float alpha[8];
      __bf16* Pw = ldsP[w];
#pragma unroll
      for (int r = 0; r < 8; ++r) {
        const int qg = qBase + r + half * 8;
#pragma unroll
        for (int ns = 0; ns < 2; ++ns) {
          const int kg = kBase + ns * 16 + l15;
          float v = s[ns][r] * scale;
          if (kg > qg) v = -1e30f;
          s[ns][r] = v;
        }
        const float tmax = xmax16(fmaxf(s[0][r], s[1][r]));
        const float mn   = fmaxf(mrow[r], tmax);
        alpha[r] = __expf(mrow[r] - mn);
        mrow[r]  = mn;
#pragma unroll
        for (int ns = 0; ns < 2; ++ns) {
          const float p = __expf(s[ns][r] - mn);
          Pw[(r + half * 8) * 32 + ns * 16 + l15] = (__bf16)p;
        }
      }
      asm volatile("s_wait_dscnt 0x0" ::: "memory");

      // P as A-fragment (16 x 32 keys)
      v16bf pf;
      *(uint4*)&pf       = *(const uint4*)(Pw + l15 * 32 + aK);
      *((uint4*)&pf + 1) = *(const uint4*)(Pw + l15 * 32 + 16 + aK);

      // rescale accumulators; PV WMMAs + ones-WMMA for row sums
#pragma unroll
      for (int nc = 0; nc < 4; ++nc) {
        v8f a = acc[nc];
#pragma unroll
        for (int r = 0; r < 8; ++r) a[r] *= alpha[r];
        const __bf16* vt = Vt + (size_t)(nc * 16 + l15) * VT_STRIDE + bK;
        v16bf vf;
        *(uint4*)&vf       = *(const uint4*)(vt);
        *((uint4*)&vf + 1) = *(const uint4*)(vt + 16);
        acc[nc] = __builtin_amdgcn_wmma_f32_16x16x32_bf16(
            false, pf, false, vf, (short)0, a, false, false);
      }
      {
        v8f a = accL;
#pragma unroll
        for (int r = 0; r < 8; ++r) a[r] *= alpha[r];
        accL = __builtin_amdgcn_wmma_f32_16x16x32_bf16(
            false, pf, false, ones, (short)0, a, false, false);
      }
    }
    __syncthreads();   // protect Kt/Vl/Vt before next iteration's staging
  }

  // ---- normalize and store ctx (bf16, [B,S,D] with head offset) ----
#pragma unroll
  for (int nc = 0; nc < 4; ++nc) {
#pragma unroll
    for (int r = 0; r < 8; ++r) {
      const float o = acc[nc][r] / accL[r];
      Ctx[headOff + (size_t)(qBase + r + half * 8) * Dd + nc * 16 + l15] =
          (__bf16)o;
    }
  }
}

// ======================================================================
// host-side orchestration
// ======================================================================
extern "C" void kernel_launch(void* const* d_in, const int* in_sizes, int n_in,
                              void* d_out, int out_size, void* d_ws,
                              size_t ws_size, hipStream_t stream) {
  const float* q  = (const float*)d_in[0];
  const float* k  = (const float*)d_in[1];
  const float* v  = (const float*)d_in[2];
  // d_in[3] = causal tril mask, implemented analytically in-kernel
  const float* Wq = (const float*)d_in[4];
  const float* bq = (const float*)d_in[5];
  const float* Wk = (const float*)d_in[6];
  const float* bk = (const float*)d_in[7];
  const float* Wv = (const float*)d_in[8];
  const float* bv = (const float*)d_in[9];
  const float* Wo = (const float*)d_in[10];
  const float* bo = (const float*)d_in[11];

  const size_t MBy = 1u << 20;
  char* ws = (char*)d_ws;
  __bf16* qbf  = (__bf16*)(ws + 0 * MBy);
  __bf16* kbf  = (__bf16*)(ws + 16 * MBy);
  __bf16* vbf  = (__bf16*)(ws + 32 * MBy);
  __bf16* wqbf = (__bf16*)(ws + 48 * MBy);
  __bf16* wkbf = (__bf16*)(ws + 50 * MBy);
  __bf16* wvbf = (__bf16*)(ws + 52 * MBy);
  __bf16* wobf = (__bf16*)(ws + 54 * MBy);
  __bf16* Qp   = (__bf16*)(ws + 56 * MBy);
  __bf16* Kp   = (__bf16*)(ws + 72 * MBy);
  __bf16* Vp   = (__bf16*)(ws + 88 * MBy);
  __bf16* ctx  = (__bf16*)(ws + 104 * MBy);

  const int nTok = BS * Dd;       // 8,388,608
  const int nW   = Dd * Dd;       // 1,048,576
  cvt_f32_bf16<<<nTok / 1024, 256, 0, stream>>>(q, qbf, nTok);
  cvt_f32_bf16<<<nTok / 1024, 256, 0, stream>>>(k, kbf, nTok);
  cvt_f32_bf16<<<nTok / 1024, 256, 0, stream>>>(v, vbf, nTok);
  cvt_f32_bf16<<<nW / 1024, 256, 0, stream>>>(Wq, wqbf, nW);
  cvt_f32_bf16<<<nW / 1024, 256, 0, stream>>>(Wk, wkbf, nW);
  cvt_f32_bf16<<<nW / 1024, 256, 0, stream>>>(Wv, wvbf, nW);
  cvt_f32_bf16<<<nW / 1024, 256, 0, stream>>>(Wo, wobf, nW);

  const int waves  = (BS / 16) * (Dd / 64);   // 8192
  const int blocks = waves / 4;               // 2048 (128 thr = 4 waves)
  gemm_xwT<false><<<blocks, 128, 0, stream>>>(qbf, wqbf, bq, (void*)Qp, BS);
  gemm_xwT<false><<<blocks, 128, 0, stream>>>(kbf, wkbf, bk, (void*)Kp, BS);
  gemm_xwT<false><<<blocks, 128, 0, stream>>>(vbf, wvbf, bv, (void*)Vp, BS);

  const int attnBlocks = Bb * Hh * (Ss / 64); // 2048
  flash_attn<<<attnBlocks, 128, 0, stream>>>(Qp, Kp, Vp, ctx);

  gemm_xwT<true><<<blocks, 128, 0, stream>>>(ctx, wobf, bo, d_out, BS);
}
